// CausalSelfAttention_35210141892654
// MI455X (gfx1250) — matrix operand order repaired and checked by reading, not compiled, not run on hardware
//
#include <hip/hip_runtime.h>
#include <hip/hip_bf16.h>
#include <math.h>

// ---------------------------------------------------------------------------
// Types for CDNA5 WMMA (wave32): V_WMMA_F32_16X16X32_BF16
// ---------------------------------------------------------------------------
typedef __bf16 bf16_t;
typedef bf16_t bf16x16 __attribute__((ext_vector_type(16)));
typedef float  v8f     __attribute__((ext_vector_type(8)));
typedef float  f32x4   __attribute__((ext_vector_type(4)));

union FragU  { bf16x16 v; f32x4 f[2]; };
union Pack16 { bf16_t b[16]; f32x4 q[2]; };

// Two separate 16-byte chunks (A-matrix fragment: K = h*16 and h*16+8)
__device__ __forceinline__ bf16x16 frag_ld2(const bf16_t* p0, const bf16_t* p1) {
  FragU u;
  u.f[0] = *(const f32x4*)p0;
  u.f[1] = *(const f32x4*)p1;
  return u.v;
}
// One contiguous 32-byte chunk (B-matrix fragment: K = h*16 .. h*16+15)
__device__ __forceinline__ bf16x16 frag_ld1(const bf16_t* p) {
  FragU u;
  u.f[0] = ((const f32x4*)p)[0];
  u.f[1] = ((const f32x4*)p)[1];
  return u.v;
}

__device__ __forceinline__ v8f wmma_bf16(bf16x16 a, bf16x16 b, v8f c) {
  return __builtin_amdgcn_wmma_f32_16x16x32_bf16(false, a, false, b, (short)0, c,
                                                 false, false);
}

// ---- CDNA5 async global->LDS copy (16B per lane), tracked by ASYNCcnt ------
// VDST operand = VGPR holding the LDS byte offset (low 32 bits of a generic
// LDS address per the aperture truncation rule), VADDR = 64-bit global addr.
__device__ __forceinline__ void async_ld16(void* lds, const void* gaddr) {
  const unsigned int      loff = (unsigned int)(unsigned long long)(uintptr_t)lds;
  const unsigned long long ga  = (unsigned long long)(uintptr_t)gaddr;
  asm volatile("global_load_async_to_lds_b128 %0, %1, off"
               :: "v"(loff), "v"(ga) : "memory");
}
__device__ __forceinline__ void wait_async0() {
  asm volatile("s_wait_asynccnt 0" ::: "memory");
}

#define D_MODEL  1024
#define N_HEADS  16
#define HEAD_DIM 64
#define SEQ      2048
#define BATCH    2
#define ROWS     (BATCH * SEQ)      // 4096
#define QKV_N    (3 * D_MODEL)      // 3072

// ---------------------------------------------------------------------------
// Kernel 1: QKV = x @ w_qkv  (f32 -> bf16 staged GEMM)
//   block tile 128(M) x 64(N), 8 waves in 4x2, each wave 32x32 register tile
//   (4 accumulators, 4 WMMAs / K-step, 2x fragment reuse), K-steps of 32.
//   epilogue: col c -> s=c%3, d=(c/3)%64, head=c/192
//     s==0 -> Q[bh][n][d] * HEAD_DIM^-0.5 ; s==1 -> K[bh][n][d]
//     s==2 -> Vt[bh][d][n] (transposed so P@V B-fragments are contiguous)
// ---------------------------------------------------------------------------
__global__ __launch_bounds__(256)
void qkv_kernel(const float* __restrict__ x, const float* __restrict__ w,
                bf16_t* __restrict__ Q, bf16_t* __restrict__ K,
                bf16_t* __restrict__ Vt) {
  __shared__ __align__(16) bf16_t lA[128][32];  // x tile, bf16 (8 KB)
  __shared__ __align__(16) bf16_t lBt[64][32];  // w tile, transposed (4 KB)

  const int m0  = blockIdx.x * 128;
  const int c0  = blockIdx.y * 64;
  const int tid = threadIdx.x;
  const int wave = tid >> 5, lane = tid & 31;
  const int mw = wave >> 1, nw = wave & 1;      // 4 x 2 wave grid
  const int l = lane & 15, h = lane >> 4;

  v8f acc[2][2] = {};

  for (int kk = 0; kk < D_MODEL; kk += 32) {
    // --- stage A tile (128x32 f32 -> bf16), 16 elems/thread ----------------
    {
      const int row = tid >> 1, c16 = (tid & 1) * 16;
      const float* src = x + (size_t)(m0 + row) * D_MODEL + kk + c16;
      f32x4 a0 = ((const f32x4*)src)[0];
      f32x4 a1 = ((const f32x4*)src)[1];
      f32x4 a2 = ((const f32x4*)src)[2];
      f32x4 a3 = ((const f32x4*)src)[3];
      Pack16 pk;
      pk.b[0]  = (bf16_t)a0.x; pk.b[1]  = (bf16_t)a0.y;
      pk.b[2]  = (bf16_t)a0.z; pk.b[3]  = (bf16_t)a0.w;
      pk.b[4]  = (bf16_t)a1.x; pk.b[5]  = (bf16_t)a1.y;
      pk.b[6]  = (bf16_t)a1.z; pk.b[7]  = (bf16_t)a1.w;
      pk.b[8]  = (bf16_t)a2.x; pk.b[9]  = (bf16_t)a2.y;
      pk.b[10] = (bf16_t)a2.z; pk.b[11] = (bf16_t)a2.w;
      pk.b[12] = (bf16_t)a3.x; pk.b[13] = (bf16_t)a3.y;
      pk.b[14] = (bf16_t)a3.z; pk.b[15] = (bf16_t)a3.w;
      f32x4* dst = (f32x4*)&lA[row][c16];
      dst[0] = pk.q[0];
      dst[1] = pk.q[1];
      if (kk + 32 < D_MODEL) __builtin_prefetch(src + 32, 0, 3);
    }
    // --- stage Bt tile (32k x 64n, transposed), 8 elems/thread -------------
    {
      const int k = tid >> 3, c8 = (tid & 7) * 8;
      const float* src = w + (size_t)(kk + k) * QKV_N + c0 + c8;
      f32x4 v0 = ((const f32x4*)src)[0];
      f32x4 v1 = ((const f32x4*)src)[1];
      lBt[c8 + 0][k] = (bf16_t)v0.x; lBt[c8 + 1][k] = (bf16_t)v0.y;
      lBt[c8 + 2][k] = (bf16_t)v0.z; lBt[c8 + 3][k] = (bf16_t)v0.w;
      lBt[c8 + 4][k] = (bf16_t)v1.x; lBt[c8 + 5][k] = (bf16_t)v1.y;
      lBt[c8 + 6][k] = (bf16_t)v1.z; lBt[c8 + 7][k] = (bf16_t)v1.w;
      if (kk + 32 < D_MODEL) __builtin_prefetch(src + 32 * QKV_N, 0, 3);
    }
    __syncthreads();

    bf16x16 a0 = frag_ld2(&lA[mw * 32 + l][h * 16],      &lA[mw * 32 + l][h * 16 + 8]);
    bf16x16 a1 = frag_ld2(&lA[mw * 32 + 16 + l][h * 16], &lA[mw * 32 + 16 + l][h * 16 + 8]);
    bf16x16 b0 = frag_ld1(&lBt[nw * 32 + l][h * 16]);
    bf16x16 b1 = frag_ld1(&lBt[nw * 32 + 16 + l][h * 16]);
    acc[0][0] = wmma_bf16(a0, b0, acc[0][0]);
    acc[0][1] = wmma_bf16(a0, b1, acc[0][1]);
    acc[1][0] = wmma_bf16(a1, b0, acc[1][0]);
    acc[1][1] = wmma_bf16(a1, b1, acc[1][1]);
    __syncthreads();
  }

  // --- epilogue: scatter into Q / K / Vt (s,d,head uniform per lane,u) -----
#pragma unroll
  for (int u = 0; u < 2; ++u) {
    const int gc   = c0 + nw * 32 + u * 16 + l;
    const int s    = gc % 3;
    const int t2   = gc / 3;
    const int d    = t2 & (HEAD_DIM - 1);
    const int head = t2 >> 6;
#pragma unroll
    for (int t = 0; t < 2; ++t) {
#pragma unroll
      for (int r = 0; r < 8; ++r) {
        const int grow = m0 + mw * 32 + t * 16 + r + 8 * h;
        const int b = grow >> 11, n = grow & (SEQ - 1);
        const size_t bh = (size_t)(b * N_HEADS + head);
        const float v = acc[t][u][r];
        if (s == 0)      Q[(bh * SEQ + n) * HEAD_DIM + d] = (bf16_t)(v * 0.125f);
        else if (s == 1) K[(bh * SEQ + n) * HEAD_DIM + d] = (bf16_t)v;
        else             Vt[(bh * HEAD_DIM + d) * SEQ + n] = (bf16_t)v;
      }
    }
  }
}

// ---------------------------------------------------------------------------
// Kernel 2: causal flash attention per (b,h), 64 q-rows/block, 32-col kv tiles.
//   K/Vt tiles double-buffered in LDS via GLOBAL_LOAD_ASYNC_TO_LDS_B128
//   (ASYNCcnt), prefetching the next tile during compute; each staged tile is
//   consumed by 4 waves (4x global-traffic reduction).
// ---------------------------------------------------------------------------
__global__ __launch_bounds__(256)
void attn_kernel(const bf16_t* __restrict__ Q, const bf16_t* __restrict__ K,
                 const bf16_t* __restrict__ Vt, bf16_t* __restrict__ attn) {
  __shared__ __align__(16) bf16_t ldsK[2][32][64];   // 8 KB
  __shared__ __align__(16) bf16_t ldsV[2][64][32];   // 8 KB
  __shared__ __align__(16) float  s_tile[64][32];    // 8 KB
  __shared__ __align__(16) bf16_t p_tile[64][32];    // 4 KB
  __shared__ float m_row[64], l_row[64], a_row[64];

  const int q0  = blockIdx.x * 64;
  const int bh  = blockIdx.y;                 // b*16 + head
  const int tid = threadIdx.x;
  const int wave = tid >> 5, lane = tid & 31;
  const int mw = wave >> 1, nw = wave & 1;
  const int l = lane & 15, h = lane >> 4;

  const bf16_t* Qb = Q  + (size_t)bh * SEQ * HEAD_DIM;
  const bf16_t* Kb = K  + (size_t)bh * SEQ * HEAD_DIM;
  const bf16_t* Vb = Vt + (size_t)bh * HEAD_DIM * SEQ;

  // Q A-fragments for this wave's 16 rows, 2 d-steps (held in VGPRs)
  const bf16_t* qrow = Qb + (size_t)(q0 + mw * 16 + l) * HEAD_DIM;
  bf16x16 qf0 = frag_ld2(qrow + h * 16,      qrow + h * 16 + 8);
  bf16x16 qf1 = frag_ld2(qrow + 32 + h * 16, qrow + 32 + h * 16 + 8);

  // per-thread 16B slices of the staged tiles
  const int krow = tid >> 3, kcol = (tid & 7) * 8;   // K tile: 32 x 64
  const int vrow = tid >> 2, vcol = (tid & 3) * 8;   // V tile: 64 x 32

  // prologue: async-stage tiles for j = 0 into buffer 0
  async_ld16(&ldsK[0][krow][kcol], Kb + (size_t)krow * HEAD_DIM + kcol);
  async_ld16(&ldsV[0][vrow][vcol], Vb + (size_t)vrow * SEQ + vcol);

  v8f o0 = {}, o1 = {};
  if (tid < 64) { m_row[tid] = -INFINITY; l_row[tid] = 0.0f; }

  const int jmax = q0 + 64;
  for (int j = 0; j < jmax; j += 32) {
    const int cur = (j >> 5) & 1;
    wait_async0();          // this wave's staged 16B slices are in LDS
    __syncthreads();        // ... and everyone else's too

    if (j + 32 < jmax) {    // prefetch next kv tile into the other buffer
      async_ld16(&ldsK[cur ^ 1][krow][kcol],
                 Kb + (size_t)(j + 32 + krow) * HEAD_DIM + kcol);
      async_ld16(&ldsV[cur ^ 1][vrow][vcol],
                 Vb + (size_t)vrow * SEQ + (j + 32) + vcol);
    }

    // ---- S = (Q*d^-1/2) @ K^T : wave's 16x16 tile, kv cols j+nw*16+l ----
    bf16x16 kf0 = frag_ld1(&ldsK[cur][nw * 16 + l][h * 16]);
    bf16x16 kf1 = frag_ld1(&ldsK[cur][nw * 16 + l][32 + h * 16]);
    v8f sacc = {};
    sacc = wmma_bf16(qf0, kf0, sacc);
    sacc = wmma_bf16(qf1, kf1, sacc);
#pragma unroll
    for (int r = 0; r < 8; ++r)
      s_tile[mw * 16 + r + 8 * h][nw * 16 + l] = sacc[r];
    __syncthreads();

    // ---- online softmax: one thread per q-row ---------------------------
    if (tid < 64) {
      const int qg = q0 + tid;
      const float mprev = m_row[tid], lprev = l_row[tid];
      float mx = mprev;
      for (int c = 0; c < 32; ++c) {
        float v = (j + c <= qg) ? s_tile[tid][c] : -INFINITY;
        mx = fmaxf(mx, v);
      }
      const float alpha = (mprev == -INFINITY) ? 0.0f : __expf(mprev - mx);
      float lsum = 0.0f;
      for (int c = 0; c < 32; ++c) {
        float v = (j + c <= qg) ? s_tile[tid][c] : -INFINITY;
        float p = __expf(v - mx);          // masked -> exp(-inf) = 0
        p_tile[tid][c] = (bf16_t)p;
        lsum += p;
      }
      m_row[tid] = mx;
      l_row[tid] = alpha * lprev + lsum;
      a_row[tid] = alpha;
    }
    __syncthreads();

    // ---- rescale O and accumulate P @ V ---------------------------------
#pragma unroll
    for (int r = 0; r < 8; ++r) {
      const float a = a_row[mw * 16 + r + 8 * h];
      o0[r] *= a;
      o1[r] *= a;
    }
    bf16x16 pf  = frag_ld2(&p_tile[mw * 16 + l][h * 16],
                           &p_tile[mw * 16 + l][h * 16 + 8]);
    bf16x16 vf0 = frag_ld1(&ldsV[cur][nw * 32 + l][h * 16]);
    bf16x16 vf1 = frag_ld1(&ldsV[cur][nw * 32 + 16 + l][h * 16]);
    o0 = wmma_bf16(pf, vf0, o0);
    o1 = wmma_bf16(pf, vf1, o1);
    __syncthreads();   // protects s_tile/p_tile/a_row AND both lds buffers
  }

  // ---- epilogue: O / l -> attn[b][n][head*64 + d] (bf16) ------------------
  const int b = bh >> 4, head = bh & (N_HEADS - 1);
#pragma unroll
  for (int r = 0; r < 8; ++r) {
    const int row = mw * 16 + r + 8 * h;
    const float inv = 1.0f / l_row[row];
    bf16_t* dst = attn + ((size_t)b * SEQ + q0 + row) * D_MODEL
                       + head * HEAD_DIM + nw * 32;
    dst[l]      = (bf16_t)(o0[r] * inv);
    dst[16 + l] = (bf16_t)(o1[r] * inv);
  }
}

// ---------------------------------------------------------------------------
// Kernel 3: out = attn(bf16) @ w_proj(f32->bf16), f32 output.
//   block tile 128x64, wave 32x32 register tile (4 WMMAs / K-step).
//   A fragments read directly from global (bf16 row-major matches A layout);
//   B staged transposed in LDS.
// ---------------------------------------------------------------------------
__global__ __launch_bounds__(256)
void proj_kernel(const bf16_t* __restrict__ attn, const float* __restrict__ w,
                 float* __restrict__ out) {
  __shared__ __align__(16) bf16_t lBt[64][32];

  const int m0  = blockIdx.x * 128;
  const int c0  = blockIdx.y * 64;
  const int tid = threadIdx.x;
  const int wave = tid >> 5, lane = tid & 31;
  const int mw = wave >> 1, nw = wave & 1;
  const int l = lane & 15, h = lane >> 4;

  v8f acc[2][2] = {};
  const bf16_t* arow0 = attn + (size_t)(m0 + mw * 32 + l) * D_MODEL;
  const bf16_t* arow1 = attn + (size_t)(m0 + mw * 32 + 16 + l) * D_MODEL;

  for (int kk = 0; kk < D_MODEL; kk += 32) {
    {
      const int k = tid >> 3, c8 = (tid & 7) * 8;
      const float* src = w + (size_t)(kk + k) * D_MODEL + c0 + c8;
      f32x4 v0 = ((const f32x4*)src)[0];
      f32x4 v1 = ((const f32x4*)src)[1];
      lBt[c8 + 0][k] = (bf16_t)v0.x; lBt[c8 + 1][k] = (bf16_t)v0.y;
      lBt[c8 + 2][k] = (bf16_t)v0.z; lBt[c8 + 3][k] = (bf16_t)v0.w;
      lBt[c8 + 4][k] = (bf16_t)v1.x; lBt[c8 + 5][k] = (bf16_t)v1.y;
      lBt[c8 + 6][k] = (bf16_t)v1.z; lBt[c8 + 7][k] = (bf16_t)v1.w;
      if (kk + 32 < D_MODEL) __builtin_prefetch(src + 32 * D_MODEL, 0, 3);
    }
    __syncthreads();

    bf16x16 a0 = frag_ld2(arow0 + kk + h * 16, arow0 + kk + h * 16 + 8);
    bf16x16 a1 = frag_ld2(arow1 + kk + h * 16, arow1 + kk + h * 16 + 8);
    bf16x16 b0 = frag_ld1(&lBt[nw * 32 + l][h * 16]);
    bf16x16 b1 = frag_ld1(&lBt[nw * 32 + 16 + l][h * 16]);
    acc[0][0] = wmma_bf16(a0, b0, acc[0][0]);
    acc[0][1] = wmma_bf16(a0, b1, acc[0][1]);
    acc[1][0] = wmma_bf16(a1, b0, acc[1][0]);
    acc[1][1] = wmma_bf16(a1, b1, acc[1][1]);
    __syncthreads();
  }

#pragma unroll
  for (int u = 0; u < 2; ++u)
#pragma unroll
    for (int t = 0; t < 2; ++t)
#pragma unroll
      for (int r = 0; r < 8; ++r) {
        const int grow = m0 + mw * 32 + t * 16 + r + 8 * h;
        out[(size_t)grow * D_MODEL + c0 + nw * 32 + u * 16 + l] = acc[t][u][r];
      }
}

// ---------------------------------------------------------------------------
// Launch: workspace layout (bytes):
//   [0,8M)   Q  bf16 [2][16][2048][64]
//   [8,16M)  K  bf16 [2][16][2048][64]
//   [16,24M) Vt bf16 [2][16][64][2048]
//   [24,32M) attn bf16 [2][2048][1024]
// ---------------------------------------------------------------------------
extern "C" void kernel_launch(void* const* d_in, const int* in_sizes, int n_in,
                              void* d_out, int out_size, void* d_ws, size_t ws_size,
                              hipStream_t stream) {
  const float* x      = (const float*)d_in[0];
  const float* w_qkv  = (const float*)d_in[1];
  const float* w_proj = (const float*)d_in[2];
  float* out = (float*)d_out;

  char* ws = (char*)d_ws;
  bf16_t* Q    = (bf16_t*)(ws);
  bf16_t* K    = (bf16_t*)(ws + (size_t)(8  << 20));
  bf16_t* Vt   = (bf16_t*)(ws + (size_t)(16 << 20));
  bf16_t* attn = (bf16_t*)(ws + (size_t)(24 << 20));

  qkv_kernel <<<dim3(ROWS / 128, QKV_N / 64),    256, 0, stream>>>(x, w_qkv, Q, K, Vt);
  attn_kernel<<<dim3(SEQ / 64, BATCH * N_HEADS), 256, 0, stream>>>(Q, K, Vt, attn);
  proj_kernel<<<dim3(ROWS / 128, D_MODEL / 64),  256, 0, stream>>>(attn, w_proj, out);
}
